// ExportableGNNBlock_1649267441700
// MI455X (gfx1250) — compile-verified, matrix operand order checked by reading, not run on hardware
//
#include <hip/hip_runtime.h>

#define N_NODES 50000
#define FIN     64
#define HDIM    128
#define E_PER   125000

typedef __attribute__((ext_vector_type(16))) __bf16 v16bf;
typedef __attribute__((ext_vector_type(8)))  float  v8f;

// Fragment geometry: a fragment is a 16(M/N) x 32(K) bf16 tile in WMMA register
// order, stored as 32 lanes x 32 bytes (lane-contiguous). For lane l:
//   row = rt*16 + (l&15), kbase = ks*32 + (l>>4)*8,
//   element i -> k = kbase + (i&7) + ((i>>3)<<4).
struct Frag32 { float4 a, b, c, d; };

__device__ __forceinline__ Frag32 load_frag_f32(const float* __restrict__ p)
{
    Frag32 r;
    r.a = *(const float4*)(p);
    r.b = *(const float4*)(p + 4);
    r.c = *(const float4*)(p + 16);
    r.d = *(const float4*)(p + 20);
    return r;
}

__device__ __forceinline__ v16bf cvt_frag(const Frag32& f)
{
    v16bf v;
    v[0]=(__bf16)f.a.x;  v[1]=(__bf16)f.a.y;  v[2]=(__bf16)f.a.z;  v[3]=(__bf16)f.a.w;
    v[4]=(__bf16)f.b.x;  v[5]=(__bf16)f.b.y;  v[6]=(__bf16)f.b.z;  v[7]=(__bf16)f.b.w;
    v[8]=(__bf16)f.c.x;  v[9]=(__bf16)f.c.y;  v[10]=(__bf16)f.c.z; v[11]=(__bf16)f.c.w;
    v[12]=(__bf16)f.d.x; v[13]=(__bf16)f.d.y; v[14]=(__bf16)f.d.z; v[15]=(__bf16)f.d.w;
    return v;
}

// ---------------------------------------------------------------------------
// Generic f32 row-major [rows x K] -> packed bf16 WMMA fragments.
// One wave of 32 threads packs one fragment. rows %16==0, K %32==0.
// ---------------------------------------------------------------------------
__global__ void pack_frags(const float* __restrict__ A, v16bf* __restrict__ P,
                           int rows, int K)
{
    const int KS = K >> 5;
    const int totalFrag = (rows >> 4) * KS;
    int tid  = blockIdx.x * blockDim.x + threadIdx.x;
    int f    = tid >> 5;
    if (f >= totalFrag) return;                 // wave-uniform
    int lane = tid & 31;
    int rt = f / KS, ks = f - rt * KS;
    int row = rt * 16 + (lane & 15);
    int kb  = ks * 32 + (lane >> 4) * 8;
    P[(size_t)f * 32 + lane] = cvt_frag(load_frag_f32(A + (size_t)row * K + kb));
}

// Fused: out = num/(sm+1e-16) + dst_feat, packed straight to bf16 frags (K=HDIM)
__global__ void pack_combine(const float* __restrict__ num, const float* __restrict__ sm,
                             const float* __restrict__ dstf, v16bf* __restrict__ P,
                             int rows)
{
    const int KS = HDIM >> 5;
    const int totalFrag = (rows >> 4) * KS;
    int tid  = blockIdx.x * blockDim.x + threadIdx.x;
    int f    = tid >> 5;
    if (f >= totalFrag) return;
    int lane = tid & 31;
    int rt = f / KS, ks = f - rt * KS;
    size_t off = (size_t)(rt * 16 + (lane & 15)) * HDIM + ks * 32 + (lane >> 4) * 8;
    Frag32 n = load_frag_f32(num + off);
    Frag32 s = load_frag_f32(sm + off);
    Frag32 d = load_frag_f32(dstf + off);
    Frag32 o;
    o.a.x=n.a.x/(s.a.x+1e-16f)+d.a.x; o.a.y=n.a.y/(s.a.y+1e-16f)+d.a.y;
    o.a.z=n.a.z/(s.a.z+1e-16f)+d.a.z; o.a.w=n.a.w/(s.a.w+1e-16f)+d.a.w;
    o.b.x=n.b.x/(s.b.x+1e-16f)+d.b.x; o.b.y=n.b.y/(s.b.y+1e-16f)+d.b.y;
    o.b.z=n.b.z/(s.b.z+1e-16f)+d.b.z; o.b.w=n.b.w/(s.b.w+1e-16f)+d.b.w;
    o.c.x=n.c.x/(s.c.x+1e-16f)+d.c.x; o.c.y=n.c.y/(s.c.y+1e-16f)+d.c.y;
    o.c.z=n.c.z/(s.c.z+1e-16f)+d.c.z; o.c.w=n.c.w/(s.c.w+1e-16f)+d.c.w;
    o.d.x=n.d.x/(s.d.x+1e-16f)+d.d.x; o.d.y=n.d.y/(s.d.y+1e-16f)+d.d.y;
    o.d.z=n.d.z/(s.d.z+1e-16f)+d.d.z; o.d.w=n.d.w/(s.d.w+1e-16f)+d.d.w;
    P[(size_t)f * 32 + lane] = cvt_frag(o);
}

// Fused BN(channel=k) + ReLU + pack to bf16 frags. h is [rows x 256] raw.
__global__ void pack_bn_relu(const float* __restrict__ h, const float* __restrict__ stats,
                             const float* __restrict__ gamma, const float* __restrict__ beta,
                             v16bf* __restrict__ P, int rows)
{
    const int KS = 256 >> 5;
    const int totalFrag = (rows >> 4) * KS;
    int tid  = blockIdx.x * blockDim.x + threadIdx.x;
    int f    = tid >> 5;
    if (f >= totalFrag) return;
    int lane = tid & 31;
    int rt = f / KS, ks = f - rt * KS;
    int kb = ks * 32 + (lane >> 4) * 8;
    Frag32 v  = load_frag_f32(h + (size_t)(rt * 16 + (lane & 15)) * 256 + kb);
    Frag32 su = load_frag_f32(stats + kb);
    Frag32 sq = load_frag_f32(stats + 256 + kb);
    Frag32 g  = load_frag_f32(gamma + kb);
    Frag32 b  = load_frag_f32(beta + kb);
    const float inv = 1.f / (float)rows;
    float* vp = (float*)&v; const float* sup = (const float*)&su;
    const float* sqp = (const float*)&sq; const float* gp = (const float*)&g;
    const float* bp = (const float*)&b;
#pragma unroll
    for (int i = 0; i < 16; ++i) {
        float mu  = sup[i] * inv;
        float var = sqp[i] * inv - mu * mu;
        float x = (vp[i] - mu) * rsqrtf(var + 1e-5f) * gp[i] + bp[i];
        vp[i] = fmaxf(x, 0.f);
    }
    P[(size_t)f * 32 + lane] = cvt_frag(v);
}

// ---------------------------------------------------------------------------
// GEMM on packed bf16 fragments: C[M x Nc] = X @ W^T (ACC: C += ...).
// One wave32 -> 16x64 slab: 1 A-frag + 4 B-frags per K-step, 4 WMMAs.
// Inner loop: pure b128 loads + V_WMMA_F32_16X16X32_BF16 (no conversions).
// ---------------------------------------------------------------------------
template<int K, bool ACC>
__global__ void gemm_packed_wmma(const v16bf* __restrict__ Xp, const v16bf* __restrict__ Wp,
                                 float* __restrict__ C, int M, int Nc)
{
    constexpr int KS = K >> 5;
    const int lane = threadIdx.x & 31;
    const int wave = blockIdx.x * (blockDim.x >> 5) + (threadIdx.x >> 5);
    const int groupsN = Nc >> 6;
    const int totalGroups = (M >> 4) * groupsN;
    if (wave >= totalGroups) return;            // wave-uniform
    const int tile_m = wave / groupsN;
    const int g_n    = wave - tile_m * groupsN;

    const v16bf* __restrict__ xf  = Xp + ((size_t)tile_m * KS) * 32 + lane;
    const v16bf* __restrict__ wf0 = Wp + ((size_t)(g_n * 4 + 0) * KS) * 32 + lane;
    const v16bf* __restrict__ wf1 = Wp + ((size_t)(g_n * 4 + 1) * KS) * 32 + lane;
    const v16bf* __restrict__ wf2 = Wp + ((size_t)(g_n * 4 + 2) * KS) * 32 + lane;
    const v16bf* __restrict__ wf3 = Wp + ((size_t)(g_n * 4 + 3) * KS) * 32 + lane;

    v8f acc0 = {}, acc1 = {}, acc2 = {}, acc3 = {};
#pragma unroll
    for (int ks = 0; ks < KS; ++ks) {
        const v16bf av = xf [(size_t)ks * 32];
        const v16bf b0 = wf0[(size_t)ks * 32];
        const v16bf b1 = wf1[(size_t)ks * 32];
        const v16bf b2 = wf2[(size_t)ks * 32];
        const v16bf b3 = wf3[(size_t)ks * 32];
        acc0 = __builtin_amdgcn_wmma_f32_16x16x32_bf16(false, av, false, b0,
                                                       (short)0, acc0, false, false);
        acc1 = __builtin_amdgcn_wmma_f32_16x16x32_bf16(false, av, false, b1,
                                                       (short)0, acc1, false, false);
        acc2 = __builtin_amdgcn_wmma_f32_16x16x32_bf16(false, av, false, b2,
                                                       (short)0, acc2, false, false);
        acc3 = __builtin_amdgcn_wmma_f32_16x16x32_bf16(false, av, false, b3,
                                                       (short)0, acc3, false, false);
    }

    float* __restrict__ cptr =
        C + (size_t)(tile_m * 16 + (lane >> 4) * 8) * Nc + g_n * 64 + (lane & 15);
#pragma unroll
    for (int j = 0; j < 8; ++j) {
        float* p = cptr + (size_t)j * Nc;
        if (ACC) {
            p[0] += acc0[j]; p[16] += acc1[j]; p[32] += acc2[j]; p[48] += acc3[j];
        } else {
            p[0]  = acc0[j]; p[16]  = acc1[j]; p[32]  = acc2[j]; p[48]  = acc3[j];
        }
    }
}

// ---------------------------------------------------------------------------
// Edge pass 1: msg = relu(x_src + ea*We) + 1e-7 ; store msg ; segment-max via
// atomicMax on uint bits (msg > 0 so float order == uint order).
// ---------------------------------------------------------------------------
__global__ void edge_msg_max(const float* __restrict__ featS,
                             const int* __restrict__ src, const int* __restrict__ dst,
                             const float* __restrict__ ea, const float* __restrict__ We,
                             float* __restrict__ msg, unsigned* __restrict__ mx, int E)
{
    const int per = HDIM / 4;
    int idx = blockIdx.x * blockDim.x + threadIdx.x;
    if (idx >= E * per) return;
    int e  = idx / per;
    int c4 = (idx - e * per) << 2;
    int s = src[e], d = dst[e];
    float eav = ea[e];
    float4 x = *(const float4*)(featS + (size_t)s * HDIM + c4);
    float4 w = *(const float4*)(We + c4);
    float4 m;
    m.x = fmaxf(x.x + eav * w.x, 0.f) + 1e-7f;
    m.y = fmaxf(x.y + eav * w.y, 0.f) + 1e-7f;
    m.z = fmaxf(x.z + eav * w.z, 0.f) + 1e-7f;
    m.w = fmaxf(x.w + eav * w.w, 0.f) + 1e-7f;
    *(float4*)(msg + (size_t)e * HDIM + c4) = m;
    unsigned* mp = mx + (size_t)d * HDIM + c4;
    atomicMax(mp + 0, __float_as_uint(m.x));
    atomicMax(mp + 1, __float_as_uint(m.y));
    atomicMax(mp + 2, __float_as_uint(m.z));
    atomicMax(mp + 3, __float_as_uint(m.w));
}

// Edge pass 2: ex = exp(msg - mx[dst]); sm += ex; num += msg*ex.
__global__ void edge_softmax_acc(const float* __restrict__ msg, const int* __restrict__ dst,
                                 const float* __restrict__ mx, float* __restrict__ sm,
                                 float* __restrict__ num, int E)
{
    const int per = HDIM / 4;
    int idx = blockIdx.x * blockDim.x + threadIdx.x;
    if (idx >= E * per) return;
    int e  = idx / per;
    int c4 = (idx - e * per) << 2;
    int d = dst[e];
    float4 m = *(const float4*)(msg + (size_t)e * HDIM + c4);
    float4 M = *(const float4*)(mx + (size_t)d * HDIM + c4);
    float ex0 = __expf(m.x - M.x), ex1 = __expf(m.y - M.y);
    float ex2 = __expf(m.z - M.z), ex3 = __expf(m.w - M.w);
    float* sp = sm  + (size_t)d * HDIM + c4;
    float* np = num + (size_t)d * HDIM + c4;
    atomicAdd(sp + 0, ex0); atomicAdd(sp + 1, ex1);
    atomicAdd(sp + 2, ex2); atomicAdd(sp + 3, ex3);
    atomicAdd(np + 0, m.x * ex0); atomicAdd(np + 1, m.y * ex1);
    atomicAdd(np + 2, m.z * ex2); atomicAdd(np + 3, m.w * ex3);
}

// BatchNorm stats: stats[0..255]=sum, stats[256..511]=sumsq
__global__ void bn_stats(const float* __restrict__ h, float* __restrict__ stats,
                         int rows, int rowsPerBlock)
{
    int c  = threadIdx.x;                 // blockDim.x == 256
    int r0 = blockIdx.x * rowsPerBlock;
    int r1 = r0 + rowsPerBlock; if (r1 > rows) r1 = rows;
    float s = 0.f, q = 0.f;
    for (int r = r0; r < r1; ++r) {
        float v = h[(size_t)r * 256 + c];
        s += v; q += v * v;
    }
    atomicAdd(&stats[c], s);
    atomicAdd(&stats[256 + c], q);
}

__global__ void leaky_relu_k(const float* __restrict__ y, float* __restrict__ out, int total)
{
    int i = blockIdx.x * blockDim.x + threadIdx.x;
    if (i >= total) return;
    float v = y[i];
    out[i] = v > 0.f ? v : 0.01f * v;
}

// ---------------------------------------------------------------------------
extern "C" void kernel_launch(void* const* d_in, const int* in_sizes, int n_in,
                              void* d_out, int out_size, void* d_ws, size_t ws_size,
                              hipStream_t stream)
{
    (void)in_sizes; (void)n_in; (void)out_size; (void)ws_size;
    const float* x_hex = (const float*)d_in[0];
    const int*   ei    = (const int*)d_in[1];
    const float* ea    = (const float*)d_in[2];
    // d_in[3] = lengths (int64, constant E_PER) — hardcoded.

    char* ws = (char*)d_ws;
    const size_t SNF  = (size_t)N_NODES * HDIM * sizeof(float);   // 25.6 MB
    const size_t SMSG = (size_t)E_PER * HDIM * sizeof(float);     // 64 MB
    const size_t SH   = (size_t)N_NODES * 256 * sizeof(float);    // 51.2 MB
    float* src_feat = (float*)(ws);
    float* dst_feat = (float*)(ws + SNF);
    float* msg      = (float*)(ws + 2 * SNF);
    float* mx       = (float*)(ws + 2 * SNF + SMSG);              // mx,sm,num contiguous
    float* sm       = (float*)(ws + 3 * SNF + SMSG);
    float* num      = (float*)(ws + 4 * SNF + SMSG);
    float* h        = (float*)(ws + 5 * SNF + SMSG);
    float* ybuf     = (float*)(ws + 5 * SNF + SMSG + SH);
    float* xcur     = (float*)(ws + 6 * SNF + SMSG + SH);
    char*  p7       = ws + 7 * SNF + SMSG + SH;
    float* stats    = (float*)(p7);                               // 2 KB
    v16bf* xhexp    = (v16bf*)(p7 + 4096);                        // 6.4 MB
    v16bf* packtmp  = (v16bf*)(p7 + 4096 + (size_t)(N_NODES/16) * 2 * 1024);   // 25.6 MB
    v16bf* w1p      = (v16bf*)(p7 + 4096 + (size_t)(N_NODES/16) * 10 * 1024);  // 64 KB
    v16bf* w2p      = (v16bf*)((char*)w1p + 64 * 1024);
    v16bf* wsp      = (v16bf*)((char*)w1p + 128 * 1024);
    v16bf* wdp      = (v16bf*)((char*)w1p + 160 * 1024);

    auto packAB = [&](const float* A, v16bf* P, int rows, int K) {
        int frags = (rows >> 4) * (K >> 5);
        pack_frags<<<(frags * 32 + 255) / 256, 256, 0, stream>>>(A, P, rows, K);
    };

    const int tE = E_PER * (HDIM / 4);
    const int bE = (tE + 255) / 256;

    packAB(x_hex, xhexp, N_NODES, FIN);          // layer-0 input packed once

    for (int layer = 0; layer < 2; ++layer) {
        float* y = (layer == 0) ? ybuf : (float*)d_out;
        hipMemsetAsync(y, 0, SNF, stream);

        for (int l = 0; l < 4; ++l) {
            // Pytree leaf order (dict keys sorted):
            const float *W1, *W2, *We, *betap, *gammap, *Wdst = nullptr, *Wsrc = nullptr;
            if (layer == 0) {                    // {W1,W2,Wdst,We,Wsrc,beta,gamma}
                int b = 4 + l * 7;
                W1     = (const float*)d_in[b + 0];
                W2     = (const float*)d_in[b + 1];
                Wdst   = (const float*)d_in[b + 2];
                We     = (const float*)d_in[b + 3];
                Wsrc   = (const float*)d_in[b + 4];
                betap  = (const float*)d_in[b + 5];
                gammap = (const float*)d_in[b + 6];
            } else {                             // {W1,W2,We,beta,gamma}
                int b = 32 + l * 5;
                W1     = (const float*)d_in[b + 0];
                W2     = (const float*)d_in[b + 1];
                We     = (const float*)d_in[b + 2];
                betap  = (const float*)d_in[b + 3];
                gammap = (const float*)d_in[b + 4];
            }

            const float *fS, *fD;
            if (layer == 0) {
                packAB(Wsrc, wsp, HDIM, FIN);
                packAB(Wdst, wdp, HDIM, FIN);
                {
                    int waves  = (N_NODES >> 4) * (HDIM >> 6);
                    int blocks = (waves * 32 + 255) / 256;
                    gemm_packed_wmma<FIN, false><<<blocks, 256, 0, stream>>>(
                        xhexp, wsp, src_feat, N_NODES, HDIM);
                    gemm_packed_wmma<FIN, false><<<blocks, 256, 0, stream>>>(
                        xhexp, wdp, dst_feat, N_NODES, HDIM);
                }
                fS = src_feat; fD = dst_feat;
            } else {
                fS = xcur; fD = xcur;
            }

            const int*   srcp = ei + l * E_PER;
            const int*   dstp = ei + 4 * E_PER + l * E_PER;
            const float* eap  = ea + l * E_PER;

            hipMemsetAsync(mx, 0, 3 * SNF, stream);   // zero mx, sm, num
            edge_msg_max<<<bE, 256, 0, stream>>>(fS, srcp, dstp, eap, We, msg,
                                                 (unsigned*)mx, E_PER);
            edge_softmax_acc<<<bE, 256, 0, stream>>>(msg, dstp, mx, sm, num, E_PER);

            // fused combine + pack (K = HDIM)
            {
                int frags = (N_NODES >> 4) * (HDIM >> 5);
                pack_combine<<<(frags * 32 + 255) / 256, 256, 0, stream>>>(
                    num, sm, fD, packtmp, N_NODES);
            }
            packAB(W1, w1p, 256, HDIM);
            {
                int waves  = (N_NODES >> 4) * (256 >> 6);
                int blocks = (waves * 32 + 255) / 256;
                gemm_packed_wmma<HDIM, false><<<blocks, 256, 0, stream>>>(
                    packtmp, w1p, h, N_NODES, 256);
            }

            hipMemsetAsync(stats, 0, 2 * 256 * sizeof(float), stream);
            const int rpb = 1024;
            bn_stats<<<(N_NODES + rpb - 1) / rpb, 256, 0, stream>>>(h, stats, N_NODES, rpb);

            // fused BN + ReLU + pack (K = 256)
            {
                int frags = (N_NODES >> 4) * (256 >> 5);
                pack_bn_relu<<<(frags * 32 + 255) / 256, 256, 0, stream>>>(
                    h, stats, gammap, betap, packtmp, N_NODES);
            }
            packAB(W2, w2p, HDIM, 256);
            {
                int waves  = (N_NODES >> 4) * (HDIM >> 6);
                int blocks = (waves * 32 + 255) / 256;
                gemm_packed_wmma<256, true><<<blocks, 256, 0, stream>>>(
                    packtmp, w2p, y, N_NODES, HDIM);
            }
        }
        if (layer == 0) {
            leaky_relu_k<<<(N_NODES * HDIM + 255) / 256, 256, 0, stream>>>(ybuf, xcur,
                                                                           N_NODES * HDIM);
        }
    }
}
